// LagrangianGNN_55173149884912
// MI455X (gfx1250) — matrix operand order repaired and checked by reading, not compiled, not run on hardware
//
#include <hip/hip_runtime.h>
#include <hip/hip_bf16.h>

#define N_PART 20000
#define HIST_N 6
#define KNBR   16
#define EEDGE  (N_PART*KNBR)
#define RADIUS 0.018f
#define VEL_STD 0.01f
#define ACC_STD 1.0f
#define EPSF   1e-8f
#define LN_EPSF 1e-5f

typedef __bf16 bf16;
typedef __attribute__((ext_vector_type(16))) __bf16 v16bf;
typedef __attribute__((ext_vector_type(8)))  float  v8f;
typedef __attribute__((ext_vector_type(4)))  unsigned int v4u;
typedef __attribute__((ext_vector_type(8)))  int    v8i;
typedef __attribute__((ext_vector_type(4)))  int    v4i;

// ---------------------------------------------------------------------------
// Feature build: [vels(10), pos(2), u_norm(2), emb(16)] -> featb bf16 [N][32]
// ---------------------------------------------------------------------------
__global__ void feat_kernel(const float* __restrict__ pos_hist,
                            const float* __restrict__ u_fluid,
                            const int*   __restrict__ ptype,
                            const float* __restrict__ type_embed,
                            bf16* __restrict__ featb,
                            float* __restrict__ posx, float* __restrict__ posy) {
  int n = blockIdx.x * blockDim.x + threadIdx.x;
  if (n >= N_PART) return;
  const float inv = 1.0f / (VEL_STD + EPSF);
  float f[32];
  const float* ph = pos_hist + (size_t)n * HIST_N * 2;
  #pragma unroll
  for (int i = 0; i < HIST_N - 1; i++) {
    f[2*i+0] = (ph[2*(i+1)+0] - ph[2*i+0]) * inv;
    f[2*i+1] = (ph[2*(i+1)+1] - ph[2*i+1]) * inv;
  }
  float px = ph[2*(HIST_N-1)+0], py = ph[2*(HIST_N-1)+1];
  f[10] = px; f[11] = py;
  f[12] = u_fluid[2*n+0] * inv;
  f[13] = u_fluid[2*n+1] * inv;
  int tp = ptype[n];
  #pragma unroll
  for (int i = 0; i < 16; i++) f[14+i] = type_embed[tp*16 + i];
  f[30] = 0.0f; f[31] = 0.0f;
  #pragma unroll
  for (int i = 0; i < 32; i++) featb[(size_t)n*32 + i] = (bf16)f[i];
  posx[n] = px; posy[n] = py;
}

// ---------------------------------------------------------------------------
// Brute-force KNN (top-16 by squared distance, self excluded)
// ---------------------------------------------------------------------------
#define KNN_CH 512
__global__ __launch_bounds__(256) void knn_kernel(const float* __restrict__ posx,
                                                  const float* __restrict__ posy,
                                                  int* __restrict__ nbr) {
  __shared__ float sx[KNN_CH], sy[KNN_CH];
  int q = blockIdx.x * 256 + threadIdx.x;
  bool active = q < N_PART;
  float qx = 0.f, qy = 0.f;
  if (active) { qx = posx[q]; qy = posy[q]; }
  float bd[KNBR]; int bi[KNBR];
  #pragma unroll
  for (int i = 0; i < KNBR; i++) { bd[i] = 3.4e38f; bi[i] = 0; }
  float worst = 3.4e38f;
  for (int base = 0; base < N_PART; base += KNN_CH) {
    int cnt = min(KNN_CH, N_PART - base);
    __syncthreads();
    for (int i = threadIdx.x; i < cnt; i += 256) { sx[i] = posx[base+i]; sy[i] = posy[base+i]; }
    __syncthreads();
    if (active) {
      for (int c = 0; c < cnt; c++) {
        float dx = sx[c] - qx, dy = sy[c] - qy;
        float d2 = dx*dx + dy*dy;
        int cid = base + c;
        if (d2 < worst && cid != q) {
          int wi = 0; float wv = bd[0];
          #pragma unroll
          for (int i = 1; i < KNBR; i++) if (bd[i] > wv) { wv = bd[i]; wi = i; }
          bd[wi] = d2; bi[wi] = cid;
          worst = bd[0];
          #pragma unroll
          for (int i = 1; i < KNBR; i++) worst = fmaxf(worst, bd[i]);
        }
      }
    }
  }
  if (active) {
    #pragma unroll
    for (int i = 0; i < KNBR; i++) nbr[(size_t)q*KNBR + i] = bi[i];
  }
}

// ---------------------------------------------------------------------------
// Edge geometry -> geomb bf16 [E][32] (cols 0..2, rest 0), vmask float [E]
// ---------------------------------------------------------------------------
__global__ void edge_geom_kernel(const float* __restrict__ posx,
                                 const float* __restrict__ posy,
                                 const int* __restrict__ nbr,
                                 bf16* __restrict__ geomb,
                                 float* __restrict__ vmask) {
  int e = blockIdx.x * blockDim.x + threadIdx.x;
  if (e >= EEDGE) return;
  int q = e / KNBR;
  int j = nbr[e];
  float dx = posx[j] - posx[q], dy = posy[j] - posy[q];
  float dist = sqrtf(dx*dx + dy*dy + 1e-12f);
  float distc = fmaxf(dist, 1e-8f);
  float tx = dx / distc, ty = dy / distc;
  float nx = -ty, ny = tx;
  float dt = dx*tx + dy*ty;
  float dn = dx*nx + dy*ny;
  bf16* p = geomb + (size_t)e * 32;
  p[0] = (bf16)(dt / RADIUS);
  p[1] = (bf16)(dn / RADIUS);
  p[2] = (bf16)(dist / RADIUS);
  #pragma unroll
  for (int i = 3; i < 32; i++) p[i] = (bf16)0.0f;
  vmask[e] = (dist < RADIUS) ? 1.0f : 0.0f;
}

__global__ void fill_zero(float* __restrict__ p, int n) {
  int i = blockIdx.x * blockDim.x + threadIdx.x;
  if (i < n) p[i] = 0.0f;
}

// ---------------------------------------------------------------------------
// Weight prep: f32 [kin][nout] -> bf16 transposed+padded [128][kpad]
// ---------------------------------------------------------------------------
__global__ void wconv_kernel(const float* __restrict__ W, int kin, int nout, int kpad,
                             bf16* __restrict__ out) {
  int i = blockIdx.x * 256 + threadIdx.x;
  if (i >= 128 * kpad) return;
  int n = i / kpad, k = i - n * kpad;
  float w = (k < kin && n < nout) ? W[(size_t)k * nout + n] : 0.0f;
  out[i] = (bf16)w;
}

// ---------------------------------------------------------------------------
// TDM: DMA a 128x32 bf16 weight tile (contiguous rows, row stride kpad elems)
// from global into LDS. Issued by one wave; completion via TENSORcnt.
// D# layout per CDNA5 ISA ch. 8 (group0: count/lds_addr/global_addr/type,
// group1: data_size/tensor dims/tile dims/stride). Groups 2/3 zero (2-D).
// ---------------------------------------------------------------------------
__device__ inline void tdm_load_wtile(const bf16* wsrc, int kpad, int ks, bf16* wt_lds) {
  unsigned lds_addr = (unsigned)(uintptr_t)wt_lds;       // flat addr low bits == LDS offset
  unsigned long long ga = (unsigned long long)(uintptr_t)(wsrc + (size_t)ks * 32);
  v4u g0;
  g0[0] = 1u;                                            // count=1, user descriptor
  g0[1] = lds_addr;                                      // lds_addr [63:32]
  g0[2] = (unsigned)(ga & 0xffffffffu);                  // global_addr [95:64]
  g0[3] = (unsigned)((ga >> 32) & 0x01ffffffu) | (2u << 30); // global_addr hi, type=2
  v8i g1;
  g1[0] = (1 << 16);                                     // wg_mask=0, data_size=1 (2 bytes)
  g1[1] = (32 & 0xffff) << 16;                           // tensor_dim0 lo16 @ [63:48]
  g1[2] = ((128 & 0xffff) << 16);                        // tensor_dim0 hi=0, tensor_dim1 lo16
  g1[3] = ((32 & 0xffff) << 16);                         // tensor_dim1 hi=0, tile_dim0=32
  g1[4] = 128;                                           // tile_dim1=128, tile_dim2=0
  g1[5] = kpad;                                          // tensor_dim0_stride lo32 (elems)
  g1[6] = 0;                                             // stride hi, dim1_stride lo
  g1[7] = 0;
  v4i z = {0,0,0,0};
#if __clang_major__ >= 23
  v8i z8 = {0,0,0,0,0,0,0,0};
  __builtin_amdgcn_tensor_load_to_lds(g0, g1, z, z, z8, 0);
#else
  __builtin_amdgcn_tensor_load_to_lds(g0, g1, z, z, 0);
#endif
}

// ---------------------------------------------------------------------------
// Fused 3-layer MLP (Kin -> 128 relu -> 128 relu -> Nout [+LN]) via bf16 WMMA
// ---------------------------------------------------------------------------
constexpr int M_NODE_ENC = 0;
constexpr int M_EDGE_ENC = 1;
constexpr int M_EDGE_BLK = 2;
constexpr int M_NODE_BLK = 3;
constexpr int M_DECODER  = 4;

struct MlpArgs {
  const bf16 *wt1, *wt2, *wt3;          // pre-transposed bf16 weights [128][kpad]
  const float *b0,*b1,*b2,*g,*beta;
  int nrows;
  const bf16 *src_a, *src_xb, *src_eb;
  const float *src_agg;
  const int *nbr;
  const float *vmask;
  float *out_x; bf16 *out_xb; bf16 *out_eb; float *out_agg; float *out_dec;
};

// A fragment (16x32 bf16): lanes 0-15 = rows, K {0..7,16..23}; lanes 16-31 K {8..15,24..31}
__device__ inline v16bf load_afrag(const bf16* base, int ld, int wave, int lane) {
  int r = lane & 15;
  int off = (lane >> 4) * 8;
  const bf16* p = base + (size_t)(wave*16 + r) * ld + off;
  v16bf a;
  *(uint4*)&a         = *(const uint4*)p;          // K off..off+7
  *(((uint4*)&a) + 1) = *(const uint4*)(p + 16);   // K off+16..off+23
  return a;
}

// B fragment (32x16 bf16) from LDS wt[n][32]: lane = column, lanes 0-15 K {0..7,16..23}; lanes 16-31 K {8..15,24..31}
__device__ inline v16bf load_bfrag(const bf16* wt, int nt, int lane) {
  int col = nt*16 + (lane & 15);
  int off = (lane >> 4) * 8;
  const bf16* p = wt + (size_t)col*32 + off;
  v16bf b;
  *(uint4*)&b         = *(const uint4*)p;
  *(((uint4*)&b) + 1) = *(const uint4*)(p + 16);
  return b;
}

template<int MODE>
__device__ inline void fill_a(const MlpArgs& A, bf16* a_lds, int row0, int ks, int t) {
  #pragma unroll
  for (int p = 0; p < 2; p++) {
    int eid = p*256 + t;              // 0..511 -> 512 chunks of 8 bf16
    int r = eid >> 2, chunk = eid & 3;
    int k = ks*32 + chunk*8;
    int grow = row0 + r;
    uint4 val = make_uint4(0u,0u,0u,0u);
    if (grow < A.nrows) {
      if constexpr (MODE == M_EDGE_BLK) {
        if (k < 128)      { int d = A.nbr[grow]; val = *(const uint4*)(A.src_xb + (size_t)d*128 + k); }
        else if (k < 256) { int s = grow >> 4;   val = *(const uint4*)(A.src_xb + (size_t)s*128 + (k-128)); }
        else              { val = *(const uint4*)(A.src_eb + (size_t)grow*128 + (k-256)); }
      } else if constexpr (MODE == M_NODE_BLK) {
        if (k < 128) val = *(const uint4*)(A.src_xb + (size_t)grow*128 + k);
        else {
          const float* ap = A.src_agg + (size_t)grow*128 + (k-128);
          __attribute__((aligned(16))) bf16 tmp[8];
          #pragma unroll
          for (int i = 0; i < 8; i++) tmp[i] = (bf16)ap[i];
          val = *(const uint4*)tmp;
        }
      } else if constexpr (MODE == M_DECODER) {
        val = *(const uint4*)(A.src_xb + (size_t)grow*128 + k);
      } else {                         // NODE_ENC / EDGE_ENC: padded ld=32 source
        val = *(const uint4*)(A.src_a + (size_t)grow*32 + k);
      }
    }
    *(uint4*)(a_lds + r*32 + chunk*8) = val;
  }
}

__device__ inline void epi_relu(v8f acc[8], const float* __restrict__ bias,
                                bf16* h, int wave, int lane) {
  int colL = lane & 15, rowadd = (lane >> 4) * 8;
  #pragma unroll
  for (int nt = 0; nt < 8; nt++) {
    float bcol = bias[nt*16 + colL];
    #pragma unroll
    for (int r = 0; r < 8; r++) {
      float v = acc[nt][r] + bcol;
      h[(size_t)(wave*16 + rowadd + r)*128 + nt*16 + colL] = (bf16)fmaxf(v, 0.0f);
      acc[nt][r] = 0.0f;
    }
  }
}

__device__ inline float redux16(float v) {
  v += __shfl_xor(v, 1, 32);
  v += __shfl_xor(v, 2, 32);
  v += __shfl_xor(v, 4, 32);
  v += __shfl_xor(v, 8, 32);
  return v;
}

template<int MODE, int KTOT>
__global__ __launch_bounds__(256) void mlp_kernel(MlpArgs A) {
  __shared__ __align__(16) bf16 a_lds[128*32];
  __shared__ __align__(16) bf16 wt_lds[128*32];
  __shared__ __align__(16) bf16 h_lds[128*128];
  const int t = threadIdx.x;
  const int wave = t >> 5, lane = t & 31;
  const int colL = lane & 15, rowadd = (lane >> 4) * 8;
  const int row0 = blockIdx.x * 128;

  v8f acc[8];
  #pragma unroll
  for (int i = 0; i < 8; i++)
    #pragma unroll
    for (int j = 0; j < 8; j++) acc[i][j] = 0.0f;

  // ---- layer 1: gathered input, TDM-staged weights ----
  #pragma unroll 1
  for (int ks = 0; ks < KTOT/32; ks++) {
    __syncthreads();                                   // prev readers of a_lds/wt_lds done
    fill_a<MODE>(A, a_lds, row0, ks, t);
    if (wave == 0) {
      tdm_load_wtile(A.wt1, KTOT, ks, wt_lds);
      __builtin_amdgcn_s_wait_tensorcnt(0);
    }
    __syncthreads();
    v16bf af = load_afrag(a_lds, 32, wave, lane);
    #pragma unroll
    for (int nt = 0; nt < 8; nt++) {
      v16bf bb = load_bfrag(wt_lds, nt, lane);
      acc[nt] = __builtin_amdgcn_wmma_f32_16x16x32_bf16(false, af, false, bb,
                                                        (short)0, acc[nt], false, false);
    }
  }
  epi_relu(acc, A.b0, h_lds, wave, lane);

  // ---- layer 2: from LDS hidden ----
  #pragma unroll 1
  for (int ks = 0; ks < 4; ks++) {
    __syncthreads();
    if (wave == 0) {
      tdm_load_wtile(A.wt2, 128, ks, wt_lds);
      __builtin_amdgcn_s_wait_tensorcnt(0);
    }
    __syncthreads();
    v16bf af = load_afrag(h_lds + ks*32, 128, wave, lane);
    #pragma unroll
    for (int nt = 0; nt < 8; nt++) {
      v16bf bb = load_bfrag(wt_lds, nt, lane);
      acc[nt] = __builtin_amdgcn_wmma_f32_16x16x32_bf16(false, af, false, bb,
                                                        (short)0, acc[nt], false, false);
    }
  }
  epi_relu(acc, A.b1, h_lds, wave, lane);

  // ---- layer 3 ----
  #pragma unroll 1
  for (int ks = 0; ks < 4; ks++) {
    __syncthreads();
    if (wave == 0) {
      tdm_load_wtile(A.wt3, 128, ks, wt_lds);
      __builtin_amdgcn_s_wait_tensorcnt(0);
    }
    __syncthreads();
    v16bf af = load_afrag(h_lds + ks*32, 128, wave, lane);
    #pragma unroll
    for (int nt = 0; nt < 8; nt++) {
      v16bf bb = load_bfrag(wt_lds, nt, lane);
      acc[nt] = __builtin_amdgcn_wmma_f32_16x16x32_bf16(false, af, false, bb,
                                                        (short)0, acc[nt], false, false);
    }
  }

  // ---- final epilogue ----
  if constexpr (MODE != M_DECODER) {
    float b2v[8], gv[8], bev[8];
    #pragma unroll
    for (int nt = 0; nt < 8; nt++) {
      int col = nt*16 + colL;
      b2v[nt] = A.b2[col]; gv[nt] = A.g[col]; bev[nt] = A.beta[col];
    }
    #pragma unroll
    for (int r = 0; r < 8; r++) {
      float s = 0.f, s2 = 0.f;
      #pragma unroll
      for (int nt = 0; nt < 8; nt++) { float v = acc[nt][r] + b2v[nt]; s += v; s2 += v*v; }
      s  = redux16(s);
      s2 = redux16(s2);
      float mu   = s  * (1.0f/128.0f);
      float var  = s2 * (1.0f/128.0f) - mu*mu;
      float rstd = rsqrtf(var + LN_EPSF);
      int grow = row0 + wave*16 + rowadd + r;
      if (grow < A.nrows) {
        float yv[8];
        #pragma unroll
        for (int nt = 0; nt < 8; nt++) {
          int col = nt*16 + colL;
          float y = (acc[nt][r] + b2v[nt] - mu) * rstd * gv[nt] + bev[nt];
          yv[nt] = y;
          if constexpr (MODE == M_NODE_ENC) {
            A.out_x[(size_t)grow*128 + col]  = y;
            A.out_xb[(size_t)grow*128 + col] = (bf16)y;
          } else if constexpr (MODE == M_EDGE_ENC || MODE == M_EDGE_BLK) {
            A.out_eb[(size_t)grow*128 + col] = (bf16)y;
          } else { // M_NODE_BLK: residual
            float xn = A.out_x[(size_t)grow*128 + col] + y;
            A.out_x[(size_t)grow*128 + col]  = xn;
            A.out_xb[(size_t)grow*128 + col] = (bf16)xn;
          }
        }
        if constexpr (MODE == M_EDGE_BLK) {
          float m = A.vmask[grow];
          if (m != 0.0f) {
            int d = A.nbr[grow];
            #pragma unroll
            for (int nt = 0; nt < 8; nt++)
              unsafeAtomicAdd(A.out_agg + (size_t)d*128 + nt*16 + colL, yv[nt] * m);
          }
        }
      }
    }
  } else {
    #pragma unroll
    for (int r = 0; r < 8; r++) {
      int grow = row0 + wave*16 + rowadd + r;
      if (grow < A.nrows && colL < 2) {
        float y = (acc[0][r] + A.b2[colL]) * (ACC_STD + EPSF);
        A.out_dec[(size_t)grow*2 + colL] = y;
      }
    }
  }
}

// ---------------------------------------------------------------------------
// Host launcher
// ---------------------------------------------------------------------------
extern "C" void kernel_launch(void* const* d_in, const int* in_sizes, int n_in,
                              void* d_out, int out_size, void* d_ws, size_t ws_size,
                              hipStream_t stream) {
  (void)in_sizes; (void)n_in; (void)out_size; (void)ws_size;
  const float* pos_hist = (const float*)d_in[0];
  const float* u_fluid  = (const float*)d_in[1];
  const int*   ptype    = (const int*)d_in[2];
  auto P = [&](int i) { return (const float*)d_in[i]; };
  // pytree flatten order: blocks[b].{edge,node} (W0,W1,W2,b0,b1,b2,beta,g),
  // decoder (W0..b2), edge_encoder, node_encoder, type_embed
  const int B_DEC = 51, B_EENC = 57, B_NENC = 65, B_TEMB = 73;

  char* w = (char*)d_ws;
  auto alloc = [&](size_t sz) -> char* { char* p = w; w += (sz + 255) & ~(size_t)255; return p; };
  bf16*  featb = (bf16*) alloc((size_t)N_PART*32*2);
  float* posx  = (float*)alloc((size_t)N_PART*4);
  float* posy  = (float*)alloc((size_t)N_PART*4);
  int*   nbr   = (int*)  alloc((size_t)EEDGE*4);
  float* vmask = (float*)alloc((size_t)EEDGE*4);
  bf16*  geomb = (bf16*) alloc((size_t)EEDGE*32*2);
  bf16*  eb    = (bf16*) alloc((size_t)EEDGE*128*2);
  bf16*  xb    = (bf16*) alloc((size_t)N_PART*128*2);
  float* x     = (float*)alloc((size_t)N_PART*128*4);
  float* agg   = (float*)alloc((size_t)N_PART*128*4);

  // --- weight prep: f32 -> transposed bf16 [128][kpad] ---
  struct WTrip { bf16 *w1, *w2, *w3; };
  auto prep = [&](int base, int kin1, int ktot, int nout3) -> WTrip {
    WTrip tr;
    tr.w1 = (bf16*)alloc((size_t)128*ktot*2);
    tr.w2 = (bf16*)alloc((size_t)128*128*2);
    tr.w3 = (bf16*)alloc((size_t)128*128*2);
    wconv_kernel<<<(128*ktot+255)/256, 256, 0, stream>>>(P(base+0), kin1, 128,  ktot, tr.w1);
    wconv_kernel<<<(128*128+255)/256, 256, 0, stream>>>(P(base+1), 128, 128,  128, tr.w2);
    wconv_kernel<<<(128*128+255)/256, 256, 0, stream>>>(P(base+2), 128, nout3, 128, tr.w3);
    return tr;
  };
  WTrip tw_nenc = prep(B_NENC, 30, 32, 128);
  WTrip tw_eenc = prep(B_EENC, 3, 32, 128);
  WTrip tw_eblk[3], tw_nblk[3];
  for (int b = 0; b < 3; b++) {
    tw_eblk[b] = prep(3 + 16*b, 384, 384, 128);
    tw_nblk[b] = prep(11 + 16*b, 256, 256, 128);
  }
  WTrip tw_dec = prep(B_DEC, 128, 128, 2);

  feat_kernel<<<(N_PART+255)/256, 256, 0, stream>>>(pos_hist, u_fluid, ptype, P(B_TEMB),
                                                    featb, posx, posy);
  knn_kernel<<<(N_PART+255)/256, 256, 0, stream>>>(posx, posy, nbr);
  edge_geom_kernel<<<(EEDGE+255)/256, 256, 0, stream>>>(posx, posy, nbr, geomb, vmask);

  auto mkargs = [&](const WTrip& tr, int base, bool ln) -> MlpArgs {
    MlpArgs a = {};
    a.wt1 = tr.w1; a.wt2 = tr.w2; a.wt3 = tr.w3;
    a.b0 = P(base+3); a.b1 = P(base+4); a.b2 = P(base+5);
    if (ln) { a.beta = P(base+6); a.g = P(base+7); }
    a.src_xb = xb; a.src_eb = eb; a.src_agg = agg;
    a.nbr = nbr; a.vmask = vmask;
    return a;
  };

  const int NT = (N_PART + 127) / 128;   // 157
  const int ET = EEDGE / 128;            // 2500

  { MlpArgs a = mkargs(tw_nenc, B_NENC, true); a.nrows = N_PART; a.src_a = featb;
    a.out_x = x; a.out_xb = xb;
    mlp_kernel<M_NODE_ENC, 32><<<NT, 256, 0, stream>>>(a); }

  { MlpArgs a = mkargs(tw_eenc, B_EENC, true); a.nrows = EEDGE; a.src_a = geomb;
    a.out_eb = eb;
    mlp_kernel<M_EDGE_ENC, 32><<<ET, 256, 0, stream>>>(a); }

  for (int b = 0; b < 3; b++) {
    fill_zero<<<(N_PART*128 + 255)/256, 256, 0, stream>>>(agg, N_PART*128);
    { MlpArgs a = mkargs(tw_eblk[b], 3 + 16*b, true); a.nrows = EEDGE;
      a.out_eb = eb; a.out_agg = agg;
      mlp_kernel<M_EDGE_BLK, 384><<<ET, 256, 0, stream>>>(a); }
    { MlpArgs a = mkargs(tw_nblk[b], 11 + 16*b, true); a.nrows = N_PART;
      a.out_x = x; a.out_xb = xb;
      mlp_kernel<M_NODE_BLK, 256><<<NT, 256, 0, stream>>>(a); }
  }

  { MlpArgs a = mkargs(tw_dec, B_DEC, false); a.nrows = N_PART;
    a.out_dec = (float*)d_out;
    mlp_kernel<M_DECODER, 128><<<NT, 256, 0, stream>>>(a); }
}
